// GeneralMaskedAttention_42545946034438
// MI455X (gfx1250) — matrix operand order repaired and checked by reading, not compiled, not run on hardware
//
#include <hip/hip_runtime.h>

// ---------------------------------------------------------------------------
// Flash-attention for gfx1250 (MI455X), wave32, v_wmma_f32_16x16x32_bf16.
// One wave (32 threads) handles one (b, h, 16-query-row) tile; 64-key blocks.
// Row-sums of P are computed by an accumulating WMMA against a ones matrix
// (replaces the ds_bpermute sum-reduction tree).
// Shapes: B=8, H=8, L=1024, D=32 (C=256).
// ---------------------------------------------------------------------------

typedef __attribute__((ext_vector_type(16))) __bf16 v16bf;
typedef __attribute__((ext_vector_type(8)))  float  v8f;

#define L_SEQ   1024
#define C_DIM   256
#define N_HEAD  8
#define D_HEAD  32
#define B_BATCH 8
#define KB      64            // keys per block iteration
#define NEG_INF (-1e9f)

__global__ __launch_bounds__(32)
void fa_gfx1250_kernel(const float* __restrict__ Q,
                       const float* __restrict__ K,
                       const float* __restrict__ V,
                       const unsigned char* __restrict__ HM,  // bool mask (B,L)
                       const float* __restrict__ BIAS,        // (B,L,L)
                       float* __restrict__ OUT)               // (B,1,L,C)
{
    // V block staged transposed (dim-major) as bf16: VT[dim][key]
    __shared__ __align__(32) __bf16 VT[32][KB];
    // P tile (16 q-rows x KB keys) bf16, row-major
    __shared__ __align__(32) __bf16 PB[16][KB];

    const int lane = threadIdx.x & 31;
    const int hh   = lane >> 4;   // lane half (0/1)
    const int ln   = lane & 15;   // lane-in-half

    const int bid = blockIdx.x;           // 0..4095
    const int qt  = bid & 63;             // q tile (L/16 = 64)
    const int h   = (bid >> 6) & 7;       // head
    const int b   = bid >> 9;             // batch
    const int q0  = qt * 16;

    const float* qbase = Q + (size_t)b * L_SEQ * C_DIM + (size_t)h * D_HEAD;
    const float* kbase = K + (size_t)b * L_SEQ * C_DIM + (size_t)h * D_HEAD;
    const float* vbase = V + (size_t)b * L_SEQ * C_DIM + (size_t)h * D_HEAD;
    const float* bbase = BIAS + ((size_t)b * L_SEQ + q0) * L_SEQ;
    const unsigned char* mbase = HM + (size_t)b * L_SEQ;

    // ---- Q A-fragment (16x32 bf16), built directly in WMMA A layout ----
    // lane<16: row=ln, K dims {0..7, 16..23}; lane>=16: row=ln, {8..15, 24..31}
    v16bf qf;
    {
        const float* qr = qbase + (size_t)(q0 + ln) * C_DIM + hh * 8;
        const float4* q4  = (const float4*)qr;         // dims hh*8 .. hh*8+7
        const float4* q4b = (const float4*)(qr + 16);  // dims 16+hh*8 ..
        float4 a0 = q4[0],  a1 = q4[1];
        float4 b0 = q4b[0], b1 = q4b[1];
        qf[0]=(__bf16)a0.x; qf[1]=(__bf16)a0.y; qf[2]=(__bf16)a0.z; qf[3]=(__bf16)a0.w;
        qf[4]=(__bf16)a1.x; qf[5]=(__bf16)a1.y; qf[6]=(__bf16)a1.z; qf[7]=(__bf16)a1.w;
        qf[8]=(__bf16)b0.x; qf[9]=(__bf16)b0.y; qf[10]=(__bf16)b0.z; qf[11]=(__bf16)b0.w;
        qf[12]=(__bf16)b1.x; qf[13]=(__bf16)b1.y; qf[14]=(__bf16)b1.z; qf[15]=(__bf16)b1.w;
    }

    // Ones B-matrix fragment (32x16 of 1.0): row-sum generator for l.
    v16bf onesf;
    #pragma unroll
    for (int i = 0; i < 16; ++i) onesf[i] = (__bf16)1.0f;

    v8f o0 = {}, o1 = {};     // output accumulators (dims 0-15, 16-31)
    v8f lacc = {};            // softmax denominator accumulator (per row)
    float mrow[8];            // running row max (rows hh*8+r)
    #pragma unroll
    for (int r = 0; r < 8; ++r) mrow[r] = -1e30f;

    const float scale = 0.17677669529663687f;   // 1/sqrt(32)

    for (int kb = 0; kb < L_SEQ; kb += KB) {
        __syncthreads();   // WAR: previous iteration's LDS reads done

        // ---- stage V block transposed into LDS: VT[dim][key] (bf16) ----
        #pragma unroll
        for (int s2 = 0; s2 < 2; ++s2) {
            const float4* v4 = (const float4*)(vbase + (size_t)(kb + s2 * 32 + lane) * C_DIM);
            #pragma unroll
            for (int d4 = 0; d4 < 8; ++d4) {
                float4 x = v4[d4];
                VT[d4*4+0][s2*32+lane] = (__bf16)x.x;
                VT[d4*4+1][s2*32+lane] = (__bf16)x.y;
                VT[d4*4+2][s2*32+lane] = (__bf16)x.z;
                VT[d4*4+3][s2*32+lane] = (__bf16)x.w;
            }
        }

        // ---- scores: four 16x16 f32 tiles via bf16 WMMA ----
        // K B-fragment: N=key ln, lane-half selects dims {0..7,16..23} / {8..15,24..31}
        v8f st[4];
        #pragma unroll
        for (int t = 0; t < 4; ++t) {
            v16bf kf;
            const float4* k4 = (const float4*)(kbase + (size_t)(kb + t * 16 + ln) * C_DIM + hh * 8);
            #pragma unroll
            for (int i = 0; i < 4; ++i) {
                float4 x = k4[(i & 1) + (i >> 1) * 4];
                kf[i*4+0]=(__bf16)x.x; kf[i*4+1]=(__bf16)x.y;
                kf[i*4+2]=(__bf16)x.z; kf[i*4+3]=(__bf16)x.w;
            }
            v8f z = {};
            st[t] = __builtin_amdgcn_wmma_f32_16x16x32_bf16(false, qf, false, kf,
                                                            (short)0, z, false, false);
        }

        bool km[4];
        #pragma unroll
        for (int t = 0; t < 4; ++t) km[t] = mbase[kb + t * 16 + ln] != 0;

        // ---- online softmax: only the row-MAX needs a cross-lane reduce ----
        float p[4][8], alpha[8];
        #pragma unroll
        for (int r = 0; r < 8; ++r) {
            const float* brow = bbase + (size_t)(r + hh * 8) * L_SEQ + kb + ln;
            float a0 = km[0] ? (st[0][r] * scale + brow[0])  : NEG_INF;
            float a1 = km[1] ? (st[1][r] * scale + brow[16]) : NEG_INF;
            float a2 = km[2] ? (st[2][r] * scale + brow[32]) : NEG_INF;
            float a3 = km[3] ? (st[3][r] * scale + brow[48]) : NEG_INF;
            float rmax = fmaxf(fmaxf(a0, a1), fmaxf(a2, a3));
            #pragma unroll
            for (int off = 1; off < 16; off <<= 1)
                rmax = fmaxf(rmax, __shfl_xor(rmax, off, 32));
            float mn = fmaxf(mrow[r], rmax);
            alpha[r] = __expf(mrow[r] - mn);
            mrow[r] = mn;
            p[0][r] = __expf(a0 - mn);
            p[1][r] = __expf(a1 - mn);
            p[2][r] = __expf(a2 - mn);
            p[3][r] = __expf(a3 - mn);
        }

        // ---- P to LDS (C-layout -> row-major), rescale accumulators ----
        #pragma unroll
        for (int r = 0; r < 8; ++r) {
            PB[r + hh * 8][ln]      = (__bf16)p[0][r];
            PB[r + hh * 8][16 + ln] = (__bf16)p[1][r];
            PB[r + hh * 8][32 + ln] = (__bf16)p[2][r];
            PB[r + hh * 8][48 + ln] = (__bf16)p[3][r];
            o0[r]   *= alpha[r];
            o1[r]   *= alpha[r];
            lacc[r] *= alpha[r];
        }

        __syncthreads();   // LDS stores (VT, PB) visible before fragment loads

        // ---- P A-fragments (16x32 each, keys 0..31 / 32..63) ----
        v16bf pf0, pf1;
        {
            const __bf16* pr = &PB[ln][hh * 8];
            #pragma unroll
            for (int i = 0; i < 8; ++i) {
                pf0[i] = pr[i];      pf0[8 + i] = pr[16 + i];
                pf1[i] = pr[32 + i]; pf1[8 + i] = pr[48 + i];
            }
        }

        // ---- P @ V : B-fragments (32 keys x 16 dims) from transposed LDS ----
        {   // output dims 0..15 (rows d=ln of VT)
            v16bf vf;
            const __bf16* va = &VT[ln][hh * 8];
            #pragma unroll
            for (int i = 0; i < 16; ++i) vf[i] = va[(i & 7) + (i >> 3) * 16];
            o0 = __builtin_amdgcn_wmma_f32_16x16x32_bf16(false, pf0, false, vf,
                                                         (short)0, o0, false, false);
            const __bf16* vb = &VT[ln][32 + hh * 8];
            #pragma unroll
            for (int i = 0; i < 16; ++i) vf[i] = vb[(i & 7) + (i >> 3) * 16];
            o0 = __builtin_amdgcn_wmma_f32_16x16x32_bf16(false, pf1, false, vf,
                                                         (short)0, o0, false, false);
        }
        {   // output dims 16..31 (rows d=16+ln of VT)
            v16bf vf;
            const __bf16* va = &VT[16 + ln][hh * 8];
            #pragma unroll
            for (int i = 0; i < 16; ++i) vf[i] = va[(i & 7) + (i >> 3) * 16];
            o1 = __builtin_amdgcn_wmma_f32_16x16x32_bf16(false, pf0, false, vf,
                                                         (short)0, o1, false, false);
            const __bf16* vb = &VT[16 + ln][32 + hh * 8];
            #pragma unroll
            for (int i = 0; i < 16; ++i) vf[i] = vb[(i & 7) + (i >> 3) * 16];
            o1 = __builtin_amdgcn_wmma_f32_16x16x32_bf16(false, pf1, false, vf,
                                                         (short)0, o1, false, false);
        }

        // ---- l += rowsum(P) via ones-matrix WMMA (no shuffle reduction) ----
        lacc = __builtin_amdgcn_wmma_f32_16x16x32_bf16(false, pf0, false, onesf,
                                                       (short)0, lacc, false, false);
        lacc = __builtin_amdgcn_wmma_f32_16x16x32_bf16(false, pf1, false, onesf,
                                                       (short)0, lacc, false, false);
    }

    // ---- epilogue: normalize, inactive-query passthrough, store ----
    float* obase = OUT + ((size_t)b * L_SEQ + q0) * C_DIM + (size_t)h * D_HEAD;
    const float* vob = vbase + (size_t)q0 * C_DIM;
    #pragma unroll
    for (int r = 0; r < 8; ++r) {
        const int m = r + hh * 8;
        const bool qactive = mbase[q0 + m] != 0;
        const float inv = 1.0f / lacc[r];
        const size_t ro = (size_t)m * C_DIM;
        float r0 = o0[r] * inv;
        float r1 = o1[r] * inv;
        if (!qactive) {
            r0 = vob[ro + ln];
            r1 = vob[ro + 16 + ln];
        }
        obase[ro + ln]      = r0;
        obase[ro + 16 + ln] = r1;
    }
}

extern "C" void kernel_launch(void* const* d_in, const int* in_sizes, int n_in,
                              void* d_out, int out_size, void* d_ws, size_t ws_size,
                              hipStream_t stream) {
    (void)in_sizes; (void)n_in; (void)out_size; (void)d_ws; (void)ws_size;
    const float* q  = (const float*)d_in[0];
    const float* k  = (const float*)d_in[1];
    const float* v  = (const float*)d_in[2];
    const unsigned char* hm = (const unsigned char*)d_in[3];  // bool mask
    const float* bias = (const float*)d_in[4];
    float* out = (float*)d_out;

    dim3 grid(B_BATCH * N_HEAD * (L_SEQ / 16));  // 4096 waves
    dim3 block(32);                               // one wave32 per block
    fa_gfx1250_kernel<<<grid, block, 0, stream>>>(q, k, v, hm, bias, out);
}
